// Siamese_39170101739813
// MI455X (gfx1250) — matrix operand order repaired and checked by reading, not compiled, not run on hardware
//
#include <hip/hip_runtime.h>

// ---------------------------------------------------------------------------
// Siamese MLP encoder on gfx1250 (MI455X, wave32), bf16x3-split WMMA GEMMs.
// fp32 ~= hi + lo (bf16 planes); GEMM accumulates hi*hi + hi*lo + lo*hi in
// fp32 via v_wmma_f32_16x16x32_bf16.  Weights pre-transposed [Npad][Kpad],
// zero-padded.  Hot loop: async global->LDS staging (double-buffered,
// ASYNCcnt) + 12 WMMAs per k-step per wave.  No guards, no conversions.
// ---------------------------------------------------------------------------

typedef __attribute__((ext_vector_type(16))) __bf16 bf16x16;
typedef __attribute__((ext_vector_type(8)))  __bf16 bf16x8;
typedef __attribute__((ext_vector_type(8)))  float  f32x8;

typedef int v4i __attribute__((vector_size(16)));
typedef __attribute__((address_space(1))) v4i as1_v4i;
typedef __attribute__((address_space(3))) v4i as3_v4i;

#define EPS_BN 1e-5f

__device__ __forceinline__ __bf16 f2bf_rn(float f) {
  unsigned u = __builtin_bit_cast(unsigned, f);
  u += 0x7FFFu + ((u >> 16) & 1u);
  unsigned short h = (unsigned short)(u >> 16);
  return __builtin_bit_cast(__bf16, h);
}
__device__ __forceinline__ float bf2f(__bf16 h) {
  unsigned short s = __builtin_bit_cast(unsigned short, h);
  unsigned u = ((unsigned)s) << 16;
  return __builtin_bit_cast(float, u);
}

// async 16B global -> LDS copy (per lane), tracked by ASYNCcnt
__device__ __forceinline__ void async_ld16(const __bf16* g, __bf16* l) {
  __builtin_amdgcn_global_load_async_to_lds_b128(
      (as1_v4i*)g, (as3_v4i*)l, 0, 0);
}

// ---- one-time: split fp32 -> bf16 hi/lo (same layout) ----
__global__ __launch_bounds__(256)
void split_f32_kernel(const float* __restrict__ X, __bf16* __restrict__ Xh,
                      __bf16* __restrict__ Xl, size_t n)
{
  size_t i = (size_t)blockIdx.x * blockDim.x + threadIdx.x;
  size_t stride = (size_t)gridDim.x * blockDim.x;
  for (; i < n; i += stride) {
    float v = X[i];
    __bf16 h = f2bf_rn(v);
    Xh[i] = h;
    Xl[i] = f2bf_rn(v - bf2f(h));
  }
}

// ---- one-time: W[K][N] fp32 -> transposed, zero-padded bf16 hi/lo planes
//      Th/Tl[Npad][Kpad] (n-major).  Grid: (Kpad/32, Npad/32), 256 thr. ----
__global__ __launch_bounds__(256)
void convert_w_kernel(const float* __restrict__ W, __bf16* __restrict__ Th,
                      __bf16* __restrict__ Tl, int K, int N, int Kpad, int Npad)
{
  __shared__ float tile[32][33];
  const int kt = blockIdx.x * 32;
  const int nt = blockIdx.y * 32;
  const int tx = threadIdx.x & 31, ty = threadIdx.x >> 5;   // 32 x 8
#pragma unroll
  for (int i = 0; i < 4; ++i) {
    int k = kt + ty + i * 8, n = nt + tx;
    tile[ty + i * 8][tx] = (k < K && n < N) ? W[(size_t)k * N + n] : 0.0f;
  }
  __syncthreads();
#pragma unroll
  for (int i = 0; i < 4; ++i) {
    int n = nt + ty + i * 8, k = kt + tx;
    float v = tile[tx][ty + i * 8];
    __bf16 h = f2bf_rn(v);
    size_t o = (size_t)n * Kpad + k;
    Th[o] = h;
    Tl[o] = f2bf_rn(v - bf2f(h));
  }
}

// ---------------------------------------------------------------------------
// GEMM: C = act(A @ W + bias [,BN]) with A [M][Kpad], W transposed [Npad][Kpad]
// Block tile 64x128, 8 waves, wave strip 16x64 -> 12 WMMAs per k-step/wave.
// Double-buffered async global->LDS staging.
// ACT: 0 = (+bias), 1 = bias+ReLU, 2 = bias+BN+ReLU
// WF32/WBF16: write fp32 plane and/or bf16 hi/lo planes (cols [N,ldc) zeroed).
// ---------------------------------------------------------------------------
template <int ACT, bool WF32, bool WBF16>
__global__ __launch_bounds__(256, 2)
void gemm_planes_kernel(const __bf16* __restrict__ Ah, const __bf16* __restrict__ Al,
                        const __bf16* __restrict__ Bh, const __bf16* __restrict__ Bl,
                        const float* __restrict__ bias,
                        const float* __restrict__ gamma, const float* __restrict__ beta,
                        const float* __restrict__ mean,  const float* __restrict__ var,
                        float* __restrict__ Cf, __bf16* __restrict__ Ch,
                        __bf16* __restrict__ Cl,
                        int M, int N, int Kpad, int ldc)
{
  (void)M;
  __shared__ __bf16 sAh[2][64][32],  sAl[2][64][32];    // 2 x 4KB each
  __shared__ __bf16 sBh[2][128][32], sBl[2][128][32];   // 2 x 8KB each

  const int tid  = threadIdx.x;
  const int lane = tid & 31;
  const int wave = tid >> 5;            // 0..7
  const int wm   = (wave & 3) * 16;     // wave row offset in 64-row tile
  const int wn   = (wave >> 2) * 64;    // wave col strip (4 x 16)
  const int m0   = blockIdx.y * 64;
  const int n0   = blockIdx.x * 128;

  f32x8 acc[4] = {};

  // staging assignments (coalesced, unguarded; everything padded)
  const int ar = tid >> 2, akc = (tid & 3) * 8;   // A: 4 lanes/row, 64B/row
  const int bn = tid >> 1, bkc = (tid & 1) * 16;  // B: 2 lanes/row, 64B/row
  const __bf16* gAh = Ah + (size_t)(m0 + ar) * Kpad + akc;
  const __bf16* gAl = Al + (size_t)(m0 + ar) * Kpad + akc;
  const __bf16* gBh = Bh + (size_t)(n0 + bn) * Kpad + bkc;
  const __bf16* gBl = Bl + (size_t)(n0 + bn) * Kpad + bkc;

  auto issue = [&](int buf, int k0) {
    async_ld16(gAh + k0,     &sAh[buf][ar][akc]);
    async_ld16(gAl + k0,     &sAl[buf][ar][akc]);
    async_ld16(gBh + k0,     &sBh[buf][bn][bkc]);
    async_ld16(gBh + k0 + 8, &sBh[buf][bn][bkc + 8]);
    async_ld16(gBl + k0,     &sBl[buf][bn][bkc]);
    async_ld16(gBl + k0 + 8, &sBl[buf][bn][bkc + 8]);
  };

  const int nk = Kpad / 32;
  issue(0, 0);                               // prologue: 6 async per wave

  for (int i = 0; i < nk; ++i) {
    const int cur = i & 1;
    if (i + 1 < nk) {
      issue(cur ^ 1, (i + 1) * 32);          // prefetch next tile (+6 in flight)
      asm volatile("s_wait_asynccnt 0x6" ::: "memory");  // cur tile done (in-order)
    } else {
      asm volatile("s_wait_asynccnt 0x0" ::: "memory");
    }
    __syncthreads();                         // all waves' async data visible

    // A fragment (16x32 bf16, contiguous K to match B: lanes<16 K 0..15, else 16..31)
    const int arow = wm + (lane & 15);
    const int kb   = (lane >> 4) * 16;
    bf16x16 ah, al;
#pragma unroll
    for (int j = 0; j < 8; ++j) {
      ah[j]     = sAh[cur][arow][kb + j];
      ah[j + 8] = sAh[cur][arow][kb + 8 + j];
      al[j]     = sAl[cur][arow][kb + j];
      al[j + 8] = sAl[cur][arow][kb + 8 + j];
    }

    const int khalf = (lane >> 4) * 16;      // B: column per lane, K half by lane>=16
#pragma unroll
    for (int t = 0; t < 4; ++t) {
      const int bc = wn + t * 16 + (lane & 15);
      bf16x16 bh = *(const bf16x16*)&sBh[cur][bc][khalf];
      bf16x16 bl = *(const bf16x16*)&sBl[cur][bc][khalf];
      acc[t] = __builtin_amdgcn_wmma_f32_16x16x32_bf16(false, ah, false, bh, (short)0, acc[t], false, false);
      acc[t] = __builtin_amdgcn_wmma_f32_16x16x32_bf16(false, ah, false, bl, (short)0, acc[t], false, false);
      acc[t] = __builtin_amdgcn_wmma_f32_16x16x32_bf16(false, al, false, bh, (short)0, acc[t], false, false);
    }
    __syncthreads();                         // safe to prefetch into buf cur next iter
  }

  // ---- epilogue (C/D layout: VGPR r -> row r + 8*(lane>=16), col = lane&15) ----
  const int rbase = m0 + wm + ((lane >> 4) * 8);
#pragma unroll
  for (int t = 0; t < 4; ++t) {
    const int col = n0 + wn + t * 16 + (lane & 15);
    if (col >= ldc) continue;
    const bool valid = col < N;
    float bia = 0.0f, sc = 1.0f, sh = 0.0f;
    if (valid) {
      if (bias) bia = bias[col];
      if (ACT == 2) {
        float inv = rsqrtf(var[col] + EPS_BN);
        sc = gamma[col] * inv;
        sh = beta[col] - mean[col] * sc;
      }
    }
#pragma unroll
    for (int r = 0; r < 8; ++r) {
      const int row = rbase + r;
      float v = 0.0f;
      if (valid) {
        v = acc[t][r] + bia;
        if (ACT == 2) v = v * sc + sh;
        if (ACT >= 1) v = fmaxf(v, 0.0f);
      }
      const size_t o = (size_t)row * ldc + col;
      if (WF32) Cf[o] = v;
      if (WBF16) {
        __bf16 h = f2bf_rn(v);
        Ch[o] = h;
        Cl[o] = f2bf_rn(v - bf2f(h));
      }
    }
  }
}

// out[i] = dot(x,y) - dot(xS,x) - dot(yS,y) + b   (one wave per row)
__global__ __launch_bounds__(256)
void siamese_reduce_kernel(const float* __restrict__ x,  const float* __restrict__ y,
                           const float* __restrict__ xS, const float* __restrict__ yS,
                           const float* __restrict__ bsc, float* __restrict__ out,
                           int rows, int D)
{
  const int lane = threadIdx.x & 31;
  const int wave = threadIdx.x >> 5;
  const int row  = blockIdx.x * (blockDim.x >> 5) + wave;
  if (row >= rows) return;

  const float* xr  = x  + (size_t)row * D;
  const float* yr  = y  + (size_t)row * D;
  const float* xSr = xS + (size_t)row * D;
  const float* ySr = yS + (size_t)row * D;

  float t = 0.0f;
  for (int j = lane; j < D; j += 32) {
    float xv = xr[j], yv = yr[j];
    t += xv * yv - xSr[j] * xv - ySr[j] * yv;
  }
#pragma unroll
  for (int off = 16; off > 0; off >>= 1) t += __shfl_xor(t, off, 32);
  if (lane == 0) out[row] = t + bsc[0];
}

extern "C" void kernel_launch(void* const* d_in, const int* in_sizes, int n_in,
                              void* d_out, int out_size, void* d_ws, size_t ws_size,
                              hipStream_t stream)
{
  (void)in_sizes; (void)n_in; (void)out_size; (void)ws_size;

  const int B    = 8192;
  const int IN   = 4096;
  const int HID  = 4096;
  const int H2   = 2048;
  const int H3   = 1365;   // HID // 3
  const int H3K  = 1376;   // padded K for layer 4 (mult of 32)
  const int H3NP = 1408;   // padded N tile coverage for layer 3 (11 * 128)
  const int OUT  = 512;

  const float* inp1 = (const float*)d_in[0];
  const float* inp2 = (const float*)d_in[1];
  const float* W1   = (const float*)d_in[2];
  const float* b1   = (const float*)d_in[3];
  const float* W2   = (const float*)d_in[4];
  const float* b2   = (const float*)d_in[5];
  const float* g1   = (const float*)d_in[6];
  const float* be1  = (const float*)d_in[7];
  const float* m1   = (const float*)d_in[8];
  const float* v1   = (const float*)d_in[9];
  const float* W3   = (const float*)d_in[10];
  const float* b3   = (const float*)d_in[11];
  const float* g2   = (const float*)d_in[12];
  const float* be2  = (const float*)d_in[13];
  const float* m2   = (const float*)d_in[14];
  const float* v2   = (const float*)d_in[15];
  const float* W4   = (const float*)d_in[16];
  const float* b4   = (const float*)d_in[17];
  const float* S    = (const float*)d_in[18];
  const float* bsc  = (const float*)d_in[19];

  // -------- workspace carve-out (256B aligned chunks) --------
  char* p = (char*)d_ws;
  auto alloc = [&](size_t bytes) -> char* {
    char* r = p;
    p += (bytes + 255) & ~(size_t)255;
    return r;
  };
  // transposed weight planes [Npad][Kpad] bf16
  __bf16* w1h = (__bf16*)alloc((size_t)HID  * IN  * 2);
  __bf16* w1l = (__bf16*)alloc((size_t)HID  * IN  * 2);
  __bf16* w2h = (__bf16*)alloc((size_t)H2   * HID * 2);
  __bf16* w2l = (__bf16*)alloc((size_t)H2   * HID * 2);
  __bf16* w3h = (__bf16*)alloc((size_t)H3NP * H2  * 2);
  __bf16* w3l = (__bf16*)alloc((size_t)H3NP * H2  * 2);
  __bf16* w4h = (__bf16*)alloc((size_t)OUT  * H3K * 2);
  __bf16* w4l = (__bf16*)alloc((size_t)OUT  * H3K * 2);
  __bf16* sth = (__bf16*)alloc((size_t)OUT  * OUT * 2);
  __bf16* stl = (__bf16*)alloc((size_t)OUT  * OUT * 2);
  // activation planes
  __bf16* inh = (__bf16*)alloc((size_t)B * IN  * 2);
  __bf16* inl = (__bf16*)alloc((size_t)B * IN  * 2);
  __bf16* h1h = (__bf16*)alloc((size_t)B * HID * 2);   // reused for h3 (1376 <= 4096)
  __bf16* h1l = (__bf16*)alloc((size_t)B * HID * 2);
  __bf16* h2h = (__bf16*)alloc((size_t)B * H2  * 2);
  __bf16* h2l = (__bf16*)alloc((size_t)B * H2  * 2);
  __bf16* xbh = (__bf16*)alloc((size_t)B * OUT * 2);
  __bf16* xbl = (__bf16*)alloc((size_t)B * OUT * 2);
  __bf16* ybh = (__bf16*)alloc((size_t)B * OUT * 2);
  __bf16* ybl = (__bf16*)alloc((size_t)B * OUT * 2);
  float*  xbf = (float*)alloc((size_t)B * OUT * 4);
  float*  ybf = (float*)alloc((size_t)B * OUT * 4);
  float*  xSf = (float*)alloc((size_t)B * OUT * 4);
  float*  ySf = (float*)alloc((size_t)B * OUT * 4);

  const dim3 blk(256);

  // -------- one-time weight split+transpose (zero-padded) --------
  convert_w_kernel<<<dim3(IN  / 32, HID  / 32), blk, 0, stream>>>(W1, w1h, w1l, IN,  HID, IN,  HID);
  convert_w_kernel<<<dim3(HID / 32, H2   / 32), blk, 0, stream>>>(W2, w2h, w2l, HID, H2,  HID, H2);
  convert_w_kernel<<<dim3(H2  / 32, H3NP / 32), blk, 0, stream>>>(W3, w3h, w3l, H2,  H3,  H2,  H3NP);
  convert_w_kernel<<<dim3(H3K / 32, OUT  / 32), blk, 0, stream>>>(W4, w4h, w4l, H3,  OUT, H3K, OUT);
  convert_w_kernel<<<dim3(OUT / 32, OUT  / 32), blk, 0, stream>>>(S,  sth, stl, OUT, OUT, OUT, OUT);

  // -------- encoders --------
  const float* inputs[2] = { inp1, inp2 };
  float*  outF[2] = { xbf, ybf };
  __bf16* outH[2] = { xbh, ybh };
  __bf16* outL[2] = { xbl, ybl };
  for (int e = 0; e < 2; ++e) {
    split_f32_kernel<<<4096, blk, 0, stream>>>(inputs[e], inh, inl, (size_t)B * IN);
    // h1 = relu(in @ W1 + b1)
    gemm_planes_kernel<1, false, true><<<dim3(HID / 128, B / 64), blk, 0, stream>>>(
        inh, inl, w1h, w1l, b1, nullptr, nullptr, nullptr, nullptr,
        nullptr, h1h, h1l, B, HID, IN, HID);
    // h2 = relu(bn1(h1 @ W2 + b2))
    gemm_planes_kernel<2, false, true><<<dim3(H2 / 128, B / 64), blk, 0, stream>>>(
        h1h, h1l, w2h, w2l, b2, g1, be1, m1, v1,
        nullptr, h2h, h2l, B, H2, HID, H2);
    // h3 = relu(bn2(h2 @ W3 + b3)), width padded to 1376 (zero-filled tail)
    gemm_planes_kernel<2, false, true><<<dim3(H3NP / 128, B / 64), blk, 0, stream>>>(
        h2h, h2l, w3h, w3l, b3, g2, be2, m2, v2,
        nullptr, h1h, h1l, B, H3, H2, H3K);
    // x = h3 @ W4 + b4  (fp32 for reduction + bf16 planes for S-GEMM)
    gemm_planes_kernel<0, true, true><<<dim3(OUT / 128, B / 64), blk, 0, stream>>>(
        h1h, h1l, w4h, w4l, b4, nullptr, nullptr, nullptr, nullptr,
        outF[e], outH[e], outL[e], B, OUT, H3K, OUT);
  }

  // -------- xS = x @ S, yS = y @ S --------
  gemm_planes_kernel<0, true, false><<<dim3(OUT / 128, B / 64), blk, 0, stream>>>(
      xbh, xbl, sth, stl, nullptr, nullptr, nullptr, nullptr, nullptr,
      xSf, nullptr, nullptr, B, OUT, OUT, OUT);
  gemm_planes_kernel<0, true, false><<<dim3(OUT / 128, B / 64), blk, 0, stream>>>(
      ybh, ybl, sth, stl, nullptr, nullptr, nullptr, nullptr, nullptr,
      ySf, nullptr, nullptr, B, OUT, OUT, OUT);

  // -------- out[i] = dot(x,y) - dot(xS,x) - dot(yS,y) + b --------
  siamese_reduce_kernel<<<dim3(B / 8), blk, 0, stream>>>(xbf, ybf, xSf, ySf, bsc,
                                                         (float*)d_out, B, OUT);
}
